// SogCLR_Loss_46411416600719
// MI455X (gfx1250) — matrix operand order repaired and checked by reading, not compiled, not run on hardware
//
#include <hip/hip_runtime.h>
#include <hip/hip_bf16.h>
#include <math.h>

#define BATCH   8192
#define EMB     256
#define NSPLIT  4
#define NT_TOTAL (BATCH / 16)        // 512 N-tiles
#define NT_PER   (NT_TOTAL / NSPLIT) // 128 N-tiles per y-block
#define SROWS   64                   // Y rows staged per LDS stage (4 N-tiles)
#define STAGE_BYTES (SROWS * EMB * 2)   // 32 KB
#define NSTAGES (NT_PER / 4)         // 32 stages per y-split
#define GAMMA_F 0.1f
#define TEMP_F  0.07f
#define EPS_F   1e-10f

typedef __bf16 bf16x4 __attribute__((ext_vector_type(4)));
typedef __bf16 bf16x8 __attribute__((ext_vector_type(8)));
typedef __bf16 v16bf  __attribute__((ext_vector_type(16)));
typedef float  v8f    __attribute__((ext_vector_type(8)));

union BFrag { v16bf v; struct { bf16x8 lo, hi; } p; };

// ---------------- f32 -> bf16 conversion (one-time, bandwidth trivial) ----------------
__global__ __launch_bounds__(256) void cvt_bf16(const float* __restrict__ src,
                                                __bf16* __restrict__ dst, int n) {
  int i = (blockIdx.x * blockDim.x + threadIdx.x) * 4;
  if (i + 3 < n) {
    float4 f = *(const float4*)(src + i);
    bf16x4 o;
    o[0] = (__bf16)f.x; o[1] = (__bf16)f.y; o[2] = (__bf16)f.z; o[3] = (__bf16)f.w;
    *(bf16x4*)(dst + i) = o;
  }
}

// ---------------- diag[i] = img[i] . txt[i], full f32 precision ----------------
__global__ __launch_bounds__(256) void diag_kernel(const float* __restrict__ img,
                                                   const float* __restrict__ txt,
                                                   float* __restrict__ diag) {
  const int wave = threadIdx.x >> 5, lane = threadIdx.x & 31;
  const int row = blockIdx.x * 8 + wave;
  const float* a = img + (size_t)row * EMB;
  const float* b = txt + (size_t)row * EMB;
  float s = 0.f;
  for (int k = lane; k < EMB; k += 32) s = fmaf(a[k], b[k], s);
  for (int m = 16; m; m >>= 1) s += __shfl_xor(s, m, 32);
  if (lane == 0) diag[row] = s;
}

// Issue one 32KB stage: 256 threads x 128B (8 x b128) via async global->LDS DMA.
// GVS addressing: SGPR base + 32-bit VGPR offset. Tracked by ASYNCcnt.
__device__ __forceinline__ void stage_issue(const char* gbase, unsigned ldsbase, int tid) {
#pragma unroll
  for (int c = 0; c < 8; ++c) {
    const unsigned voff  = (unsigned)tid * 128u + (unsigned)c * 16u;
    const unsigned laddr = ldsbase + voff;
    asm volatile("global_load_async_to_lds_b128 %0, %1, %2"
                 :: "v"(laddr), "v"(voff), "s"(gbase) : "memory");
  }
}

// ---------------- fused GEMM + row statistics, LDS-staged B ----------------
// C = X @ Y^T in 16x16x32 bf16 WMMA tiles; B panels staged via async DMA to LDS,
// double-buffered; A fragments live in VGPRs for the whole N sweep.
// MODE 0: out0[y*B + i] = max_j C[i][j]                 (partial over this y-split)
// MODE 1: d = (C[i][j] - diag[i]) / T;  e = exp(d - nb[i]);  out0 += e, out1 += e*d
template<int MODE>
__global__ __launch_bounds__(256)
void gemm_rowstat(const __bf16* __restrict__ Xb, const __bf16* __restrict__ Yb,
                  const float* __restrict__ diag, const float* __restrict__ nb,
                  float* __restrict__ out0, float* __restrict__ out1) {
  __shared__ __align__(16) char smem[2][STAGE_BYTES];   // 64 KB double buffer

  const int tid  = threadIdx.x;
  const int wave = tid >> 5;
  const int lane = tid & 31;
  const int half = lane >> 4;       // 0: lanes 0-15, 1: lanes 16-31
  const int l16  = lane & 15;
  const int rowBase = blockIdx.x * 128 + wave * 16;

  // Preload all A fragments for this wave's 16 rows (reused across the whole N sweep).
  // 16-bit A 16x32 layout: per lane, K chunks {h*8..h*8+7} and {16+h*8..23+h*8}.
  BFrag a[8];
  const __bf16* arow = Xb + (size_t)(rowBase + l16) * EMB;
#pragma unroll
  for (int kt = 0; kt < 8; ++kt) {
    const int k0 = kt * 32 + half * 8;
    a[kt].p.lo = *(const bf16x8*)(arow + k0);
    a[kt].p.hi = *(const bf16x8*)(arow + k0 + 16);
  }

  float st0[8], st1[8], dv[8], nbv[8];
#pragma unroll
  for (int r = 0; r < 8; ++r) { st0[r] = (MODE == 0) ? -__builtin_inff() : 0.f; st1[r] = 0.f; }
  if (MODE == 1) {
#pragma unroll
    for (int r = 0; r < 8; ++r) {
      dv[r]  = diag[rowBase + half * 8 + r];   // C rows per lane: M = half*8 + r
      nbv[r] = nb  [rowBase + half * 8 + r];
    }
  }

  const float invT = 1.0f / TEMP_F;
  const char* ybase = (const char*)(Yb + (size_t)(blockIdx.y * NT_PER * 16) * EMB);
  const unsigned lds0 = (unsigned)(uintptr_t)(&smem[0][0]);
  const unsigned lds1 = (unsigned)(uintptr_t)(&smem[1][0]);

  stage_issue(ybase, lds0, tid);   // prologue: DMA stage 0

  for (int s = 0; s < NSTAGES; ++s) {
    const int buf = s & 1;
    if (s + 1 < NSTAGES) {
      stage_issue(ybase + (size_t)(s + 1) * STAGE_BYTES, buf ? lds0 : lds1, tid);
      // in-order async completion: <=8 outstanding means current stage's 8 are done
      asm volatile("s_wait_asynccnt 0x8" ::: "memory");
    } else {
      asm volatile("s_wait_asynccnt 0x0" ::: "memory");
    }
    __syncthreads();

    const char* sbuf = smem[buf];
#pragma unroll
    for (int t = 0; t < 4; ++t) {
      // 16-bit B 32x16 layout: lane n holds column n, K chunks {h*8..h*8+7} and {16+h*8..23+h*8}.
      const __bf16* brow = (const __bf16*)sbuf + (size_t)(t * 16 + l16) * EMB;
      v8f c0 = {}, c1 = {};
#pragma unroll
      for (int kt = 0; kt < 8; kt += 2) {
        BFrag b0, b1;
        const int ka = kt * 32 + half * 8;
        const int kb = (kt + 1) * 32 + half * 8;
        b0.p.lo = *(const bf16x8*)(brow + ka);  b0.p.hi = *(const bf16x8*)(brow + ka + 16);
        b1.p.lo = *(const bf16x8*)(brow + kb);  b1.p.hi = *(const bf16x8*)(brow + kb + 16);
        c0 = __builtin_amdgcn_wmma_f32_16x16x32_bf16(false, a[kt].v,     false, b0.v,
                                                     (short)0, c0, false, false);
        c1 = __builtin_amdgcn_wmma_f32_16x16x32_bf16(false, a[kt + 1].v, false, b1.v,
                                                     (short)0, c1, false, false);
      }
#pragma unroll
      for (int r = 0; r < 8; ++r) {
        const float v = c0[r] + c1[r];
        if (MODE == 0) {
          st0[r] = fmaxf(st0[r], v);
        } else {
          const float d = (v - dv[r]) * invT;
          const float e = __expf(d - nbv[r]);
          st0[r] += e;
          st1[r] = fmaf(e, d, st1[r]);
        }
      }
    }
    __syncthreads();   // all waves done reading buf before it is DMA-overwritten
  }

  // Reduce across the 16 lanes holding the same row (C layout: N = lane&15).
#pragma unroll
  for (int r = 0; r < 8; ++r) {
#pragma unroll
    for (int m = 1; m <= 8; m <<= 1) {
      if (MODE == 0) {
        st0[r] = fmaxf(st0[r], __shfl_xor(st0[r], m, 32));
      } else {
        st0[r] += __shfl_xor(st0[r], m, 32);
        st1[r] += __shfl_xor(st1[r], m, 32);
      }
    }
  }
  if (l16 == 0) {
    const int base = blockIdx.y * BATCH + rowBase + half * 8;
#pragma unroll
    for (int r = 0; r < 8; ++r) {
      out0[base + r] = st0[r];
      if (MODE == 1) out1[base + r] = st1[r];
    }
  }
}

// ---------------- combine row-max partials, form new_b for both sides ----------------
__global__ __launch_bounds__(256)
void newb_kernel(const float* __restrict__ rmI, const float* __restrict__ rmT,
                 const float* __restrict__ diag,
                 const float* __restrict__ b_I, const float* __restrict__ b_T,
                 const int* __restrict__ idsI, const int* __restrict__ idsT,
                 float* __restrict__ nbI, float* __restrict__ nbT) {
  const int i = blockIdx.x * blockDim.x + threadIdx.x;
  if (i >= BATCH) return;
  float mI = -__builtin_inff(), mT = -__builtin_inff();
  for (int s = 0; s < NSPLIT; ++s) {
    mI = fmaxf(mI, rmI[s * BATCH + i]);
    mT = fmaxf(mT, rmT[s * BATCH + i]);
  }
  const float d = diag[i];
  nbI[i] = fmaxf((mI - d) / TEMP_F, b_I[idsI[i]]);
  nbT[i] = fmaxf((mT - d) / TEMP_F, b_T[idsT[i]]);
}

// ---------------- final scalar: EMA normalizer + weighted loss, single block ----------------
__global__ __launch_bounds__(1024)
void finalize_kernel(const float* __restrict__ S0I, const float* __restrict__ S1I,
                     const float* __restrict__ S0T, const float* __restrict__ S1T,
                     const float* __restrict__ nbI, const float* __restrict__ nbT,
                     const float* __restrict__ b_I, const float* __restrict__ b_T,
                     const float* __restrict__ s_I, const float* __restrict__ s_T,
                     const int* __restrict__ idsI, const int* __restrict__ idsT,
                     float* __restrict__ out) {
  __shared__ float red[1024];
  const float invBm1 = 1.0f / (float)(BATCH - 1);
  float acc = 0.f;
  for (int i = threadIdx.x; i < BATCH; i += 1024) {
    {
      float s0 = 0.f, s1 = 0.f;
      for (int s = 0; s < NSPLIT; ++s) { s0 += S0I[s * BATCH + i]; s1 += S1I[s * BATCH + i]; }
      const float nb = nbI[i], ob = b_I[idsI[i]];
      const float g  = s0 * invBm1;
      const float sn = (1.f - GAMMA_F) * s_I[idsI[i]] * __expf(ob - nb) + GAMMA_F * g;
      acc += TEMP_F * s1 / (sn + EPS_F) * invBm1;
    }
    {
      float s0 = 0.f, s1 = 0.f;
      for (int s = 0; s < NSPLIT; ++s) { s0 += S0T[s * BATCH + i]; s1 += S1T[s * BATCH + i]; }
      const float nb = nbT[i], ob = b_T[idsT[i]];
      const float g  = s0 * invBm1;
      const float sn = (1.f - GAMMA_F) * s_T[idsT[i]] * __expf(ob - nb) + GAMMA_F * g;
      acc += TEMP_F * s1 / (sn + EPS_F) * invBm1;
    }
  }
  red[threadIdx.x] = acc;
  __syncthreads();
  for (int off = 512; off; off >>= 1) {
    if (threadIdx.x < off) red[threadIdx.x] += red[threadIdx.x + off];
    __syncthreads();
  }
  if (threadIdx.x == 0) out[0] = red[0] * (1.0f / (float)BATCH);
}

extern "C" void kernel_launch(void* const* d_in, const int* in_sizes, int n_in,
                              void* d_out, int out_size, void* d_ws, size_t ws_size,
                              hipStream_t stream) {
  const float* img  = (const float*)d_in[0];
  const float* txt  = (const float*)d_in[1];
  const float* b_I  = (const float*)d_in[2];
  const float* b_T  = (const float*)d_in[3];
  const float* s_I  = (const float*)d_in[4];
  const float* s_T  = (const float*)d_in[5];
  const int*   idsI = (const int*)d_in[6];
  const int*   idsT = (const int*)d_in[7];
  float* out = (float*)d_out;

  // workspace layout: bf16 copies (8 MB) then float scratch (~0.9 MB)
  char* ws = (char*)d_ws;
  __bf16* imgB = (__bf16*)ws;
  __bf16* txtB = (__bf16*)(ws + (size_t)BATCH * EMB * 2);
  float*  fb   = (float*)(ws + (size_t)BATCH * EMB * 4);
  float* diag = fb;
  float* rmI  = diag + BATCH;
  float* rmT  = rmI + NSPLIT * BATCH;
  float* nbI  = rmT + NSPLIT * BATCH;
  float* nbT  = nbI + BATCH;
  float* S0I  = nbT + BATCH;
  float* S1I  = S0I + NSPLIT * BATCH;
  float* S0T  = S1I + NSPLIT * BATCH;
  float* S1T  = S0T + NSPLIT * BATCH;

  const int nElem = BATCH * EMB;
  cvt_bf16<<<nElem / 4 / 256, 256, 0, stream>>>(img, imgB, nElem);
  cvt_bf16<<<nElem / 4 / 256, 256, 0, stream>>>(txt, txtB, nElem);
  diag_kernel<<<BATCH / 8, 256, 0, stream>>>(img, txt, diag);

  dim3 grid(BATCH / 128, NSPLIT);
  // Pass A: row maxes of sim (image side) and of sim^T (text side, swapped operands)
  gemm_rowstat<0><<<grid, 256, 0, stream>>>(imgB, txtB, nullptr, nullptr, rmI, nullptr);
  gemm_rowstat<0><<<grid, 256, 0, stream>>>(txtB, imgB, nullptr, nullptr, rmT, nullptr);
  newb_kernel<<<BATCH / 256, 256, 0, stream>>>(rmI, rmT, diag, b_I, b_T, idsI, idsT, nbI, nbT);
  // Pass B: sum(exp) and sum(exp * diff) per row of sim / sim^T
  gemm_rowstat<1><<<grid, 256, 0, stream>>>(imgB, txtB, diag, nbI, S0I, S1I);
  gemm_rowstat<1><<<grid, 256, 0, stream>>>(txtB, imgB, diag, nbT, S0T, S1T);
  finalize_kernel<<<1, 1024, 0, stream>>>(S0I, S1I, S0T, S1T, nbI, nbT,
                                          b_I, b_T, s_I, s_T, idsI, idsT, out);
}
